// HiERABlock_64716567216785
// MI455X (gfx1250) — matrix operand order repaired and checked by reading, not compiled, not run on hardware
//
#include <hip/hip_runtime.h>
#include <hip/hip_bf16.h>
#include <math.h>

typedef __bf16 bf16;
typedef __attribute__((ext_vector_type(8)))  __bf16 v8bf;
typedef __attribute__((ext_vector_type(16))) __bf16 v16bf;
typedef __attribute__((ext_vector_type(8)))  float  v8f;

#define WSZ 14
#define NHD 12
#define NTOK 196          // tokens per window
#define NPAD 224          // padded to 14 tiles of 16
#define CD 768
#define HD 64
#define BB 8
#define HW 56
#define MTOK (BB*HW*HW)   // 25088
#define DFF 3072
#define QKV_LD 2304
#define MPAD (MTOK + 64)  // qkv row padding for window-tile overreach

// ---------- WMMA helpers ----------
__device__ __forceinline__ v16bf ldpair(const bf16* p0, const bf16* p1) {
  v8bf a = *(const v8bf*)p0;
  v8bf b = *(const v8bf*)p1;
  return __builtin_shufflevector(a, b, 0,1,2,3,4,5,6,7,8,9,10,11,12,13,14,15);
}

__device__ __forceinline__ v8f wmma_bf16(v16bf a, v16bf b, v8f c) {
  return __builtin_amdgcn_wmma_f32_16x16x32_bf16(false, a, false, b, (short)0, c, false, false);
}

// A: (rows x K) bf16 row-major, WT: (cols x K) bf16 row-major (transposed weight)
// 32x64 output strip: two stacked 16-row tiles (accL/accH) share the same B
// operands -> 12 b128 loads per 8 WMMAs. Ping-pong pipeline, unrolled x2 so
// there are no loop-carried vector copies. Requires K/32 even (768, 3072 ok).
__device__ __forceinline__ void gemm_strip32(const bf16* __restrict__ A,
                                             const bf16* __restrict__ WT,
                                             int lda, int ldb,
                                             size_t row0, int col0, int K,
                                             v8f accL[4], v8f accH[4]) {
  int lane = threadIdx.x & 31;
  int m = lane & 15, ksel = lane >> 4;
  const bf16* arowL = A + (row0 + (size_t)m) * (size_t)lda + ksel*16;
  const bf16* arowH = arowL + (size_t)16 * lda;
  const bf16* bbase = WT + (size_t)(col0 + m) * (size_t)ldb + ksel*16;

  v16bf aL0 = ldpair(arowL, arowL + 8);
  v16bf aH0 = ldpair(arowH, arowH + 8);
  v16bf b0[4];
#pragma unroll
  for (int n = 0; n < 4; ++n) {
    const bf16* bp = bbase + (size_t)n * 16 * ldb;
    b0[n] = ldpair(bp, bp + 8);
  }
  v16bf aL1, aH1; v16bf b1[4];
  const int T = K / 32;          // even
  int koff = 32;
  for (int s = 0; s < T - 2; s += 2) {
    __builtin_prefetch(arowL + koff + 512, 0, 3);  // global_prefetch_b8
    // load odd stage
    aL1 = ldpair(arowL + koff, arowL + koff + 8);
    aH1 = ldpair(arowH + koff, arowH + koff + 8);
#pragma unroll
    for (int n = 0; n < 4; ++n) {
      const bf16* bp = bbase + (size_t)n * 16 * ldb + koff;
      b1[n] = ldpair(bp, bp + 8);
    }
    // compute even stage
#pragma unroll
    for (int n = 0; n < 4; ++n) accL[n] = wmma_bf16(aL0, b0[n], accL[n]);
#pragma unroll
    for (int n = 0; n < 4; ++n) accH[n] = wmma_bf16(aH0, b0[n], accH[n]);
    // load even stage (koff+32)
    aL0 = ldpair(arowL + koff + 32, arowL + koff + 40);
    aH0 = ldpair(arowH + koff + 32, arowH + koff + 40);
#pragma unroll
    for (int n = 0; n < 4; ++n) {
      const bf16* bp = bbase + (size_t)n * 16 * ldb + koff + 32;
      b0[n] = ldpair(bp, bp + 8);
    }
    // compute odd stage
#pragma unroll
    for (int n = 0; n < 4; ++n) accL[n] = wmma_bf16(aL1, b1[n], accL[n]);
#pragma unroll
    for (int n = 0; n < 4; ++n) accH[n] = wmma_bf16(aH1, b1[n], accH[n]);
    koff += 64;
  }
  // tail: buf0 holds step T-2; load step T-1 then compute both
  aL1 = ldpair(arowL + koff, arowL + koff + 8);
  aH1 = ldpair(arowH + koff, arowH + koff + 8);
#pragma unroll
  for (int n = 0; n < 4; ++n) {
    const bf16* bp = bbase + (size_t)n * 16 * ldb + koff;
    b1[n] = ldpair(bp, bp + 8);
  }
#pragma unroll
  for (int n = 0; n < 4; ++n) accL[n] = wmma_bf16(aL0, b0[n], accL[n]);
#pragma unroll
  for (int n = 0; n < 4; ++n) accH[n] = wmma_bf16(aH0, b0[n], accH[n]);
#pragma unroll
  for (int n = 0; n < 4; ++n) accL[n] = wmma_bf16(aL1, b1[n], accL[n]);
#pragma unroll
  for (int n = 0; n < 4; ++n) accH[n] = wmma_bf16(aH1, b1[n], accH[n]);
}

// ---------- kernel 1: f32 weight -> bf16 transposed ----------
__global__ void k_wT(const float* __restrict__ in, bf16* __restrict__ out, int K, int N) {
  int idx = blockIdx.x * blockDim.x + threadIdx.x;
  if (idx >= K * N) return;
  int k = idx / N, n = idx % N;
  out[(size_t)n * K + k] = (bf16)in[idx];
}

// ---------- kernel 2: LN1 + BCHW->token transpose ----------
// writes xs (f32, linear token order) and h (bf16, window token order)
__global__ __launch_bounds__(256) void k_ln1(const float* __restrict__ x,
                                             const float* __restrict__ w,
                                             const float* __restrict__ b,
                                             float* __restrict__ xs,
                                             bf16* __restrict__ h) {
  int wave = (blockIdx.x * blockDim.x + threadIdx.x) >> 5;
  int lane = threadIdx.x & 31;
  if (wave >= MTOK) return;
  int t = wave;
  int bi = t / 3136, rem = t % 3136;
  int y = rem / 56, xc = rem % 56;
  const float* px = x + (size_t)bi * CD * 3136 + y * 56 + xc;
  float v[24], s = 0.f, s2 = 0.f;
#pragma unroll
  for (int i = 0; i < 24; ++i) {
    float f = px[(size_t)(lane + 32*i) * 3136];
    v[i] = f; s += f; s2 += f*f;
  }
#pragma unroll
  for (int mk = 16; mk >= 1; mk >>= 1) { s += __shfl_xor(s, mk); s2 += __shfl_xor(s2, mk); }
  float mu = s * (1.f/768.f);
  float rstd = rsqrtf(s2 * (1.f/768.f) - mu*mu + 1e-5f);
  int hb = y / WSZ, i1 = y % WSZ, wb = xc / WSZ, j1 = xc % WSZ;
  size_t twin = ((size_t)((bi*4 + hb)*4 + wb))*NTOK + i1*WSZ + j1;
  float* pxs = xs + (size_t)t * CD;
  bf16*  ph  = h  + twin * CD;
#pragma unroll
  for (int i = 0; i < 24; ++i) {
    int c = lane + 32*i;
    pxs[c] = v[i];
    ph[c]  = (bf16)((v[i]-mu)*rstd * w[c] + b[c]);
  }
}

// ---------- kernel 3: QKV GEMM (32x64 strips) ----------
__global__ __launch_bounds__(256) void k_gemm_qkv(const bf16* __restrict__ h,
                                                  const bf16* __restrict__ WT,
                                                  const float* __restrict__ bias,
                                                  bf16* __restrict__ qkv) {
  int wave = (blockIdx.x * blockDim.x + threadIdx.x) >> 5;
  const int cbs = QKV_LD / 64;
  if (wave >= (MTOK/32) * cbs) return;
  int rt = wave / cbs, cb = wave % cbs;
  size_t row0 = (size_t)rt * 32; int col0 = cb * 64;
  int lane = threadIdx.x & 31, m = lane & 15, ksel = lane >> 4;
  v8f accL[4] = {}, accH[4] = {};
  gemm_strip32(h, WT, CD, CD, row0, col0, CD, accL, accH);
#pragma unroll
  for (int n = 0; n < 4; ++n) {
    int col = col0 + n*16 + m;
    float bj = bias[col];
#pragma unroll
    for (int r = 0; r < 8; ++r) {
      size_t row = row0 + r + ksel*8;
      qkv[row*QKV_LD + col]        = (bf16)(accL[n][r] + bj);
      qkv[(row + 16)*QKV_LD + col] = (bf16)(accH[n][r] + bj);
    }
  }
}

// ---------- kernel 4: windowed attention, one (head,window) per block ----------
__global__ __launch_bounds__(128) void k_attn(const bf16* __restrict__ qkv,
                                              const float* __restrict__ table,
                                              bf16* __restrict__ O) {
  __shared__ bf16 Vt[HD * NPAD];        // V transposed: [d][token]
  __shared__ bf16 P[4 * 16 * NPAD];     // per-wave softmax strip
  int head = blockIdx.x, win = blockIdx.y;
  int tid = threadIdx.x;
  int lane = tid & 31, wv = tid >> 5;
  int m = lane & 15, ksel = lane >> 4;

  for (int e = tid; e < HD * NPAD; e += 128) {
    int d = e / NPAD, tk = e % NPAD;
    bf16 val = (bf16)0.0f;
    if (tk < NTOK) val = qkv[((size_t)win*NTOK + tk)*QKV_LD + 2*CD + head*HD + d];
    Vt[d*NPAD + tk] = val;
  }
  __syncthreads();

  const bf16* qbase = qkv + (size_t)win*NTOK*QKV_LD + head*HD;
  for (int strip = wv; strip < 13; strip += 4) {
    int rbase = strip * 16;
    v8f S[14];
    // ---- S = Q K^T  (B operands double-buffered across n tiles) ----
    const bf16* arow = qbase + (size_t)(rbase + m) * QKV_LD + ksel*16;
    v16bf a0 = ldpair(arow,      arow + 8);
    v16bf a1 = ldpair(arow + 32, arow + 40);
    const bf16* kb0 = qkv + ((size_t)win*NTOK + m)*QKV_LD + CD + head*HD + ksel*16;
    v16bf bc0 = ldpair(kb0,      kb0 + 8);
    v16bf bc1 = ldpair(kb0 + 32, kb0 + 40);
#pragma unroll
    for (int n = 0; n < 13; ++n) {
      v16bf bn0, bn1;
      if (n < 12) {
        const bf16* kb = kb0 + (size_t)(n + 1) * 16 * QKV_LD;
        bn0 = ldpair(kb,      kb + 8);
        bn1 = ldpair(kb + 32, kb + 40);
      }
      v8f acc = {};
      acc = wmma_bf16(a0, bc0, acc);
      acc = wmma_bf16(a1, bc1, acc);
      S[n] = acc;
      bc0 = bn0; bc1 = bn1;
    }
    // ---- scale + rel-pos bias + mask ----
    int row[8], ri[8], rj[8];
#pragma unroll
    for (int r = 0; r < 8; ++r) {
      row[r] = rbase + r + ksel*8;
      int rr = row[r] < NTOK ? row[r] : 0;
      ri[r] = rr / WSZ; rj[r] = rr % WSZ;
    }
#pragma unroll
    for (int n = 0; n < 14; ++n) {
      int col = n*16 + m;
      int i2 = col / WSZ, j2 = col % WSZ;
#pragma unroll
      for (int r = 0; r < 8; ++r) {
        float sv;
        if (col >= NTOK)          sv = -1e30f;
        else if (row[r] >= NTOK)  sv = 0.0f;
        else {
          int idx = (ri[r]-i2+13)*27 + (rj[r]-j2+13);
          sv = S[n][r]*0.125f + table[idx*NHD + head];
        }
        S[n][r] = sv;
      }
    }
    // ---- softmax along row (rows live per 16-lane group in C layout) ----
    float rmax[8], rsum[8], rinv[8];
#pragma unroll
    for (int r = 0; r < 8; ++r) rmax[r] = -3e38f;
#pragma unroll
    for (int n = 0; n < 14; ++n)
#pragma unroll
      for (int r = 0; r < 8; ++r) rmax[r] = fmaxf(rmax[r], S[n][r]);
#pragma unroll
    for (int r = 0; r < 8; ++r) {
#pragma unroll
      for (int mk = 8; mk >= 1; mk >>= 1) rmax[r] = fmaxf(rmax[r], __shfl_xor(rmax[r], mk));
      rsum[r] = 0.f;
    }
#pragma unroll
    for (int n = 0; n < 14; ++n)
#pragma unroll
      for (int r = 0; r < 8; ++r) {
        float e = __expf(S[n][r] - rmax[r]);
        S[n][r] = e; rsum[r] += e;
      }
#pragma unroll
    for (int r = 0; r < 8; ++r) {
#pragma unroll
      for (int mk = 8; mk >= 1; mk >>= 1) rsum[r] += __shfl_xor(rsum[r], mk);
      rinv[r] = 1.0f / rsum[r];
    }
    // ---- P to LDS (bf16) ----
    bf16* Pw = P + wv * 16 * NPAD;
#pragma unroll
    for (int n = 0; n < 14; ++n) {
      int col = n*16 + m;
#pragma unroll
      for (int r = 0; r < 8; ++r)
        Pw[(r + ksel*8)*NPAD + col] = (bf16)(S[n][r] * rinv[r]);
    }
    // ---- O = P V  (LDS operands double-buffered across K steps) ----
    v8f Oa[4] = {};
    const bf16* prow = Pw + m * NPAD + ksel*16;
    const bf16* vb0  = Vt + m * NPAD + ksel*16;
    v16bf ac = ldpair(prow, prow + 8);
    v16bf bc[4];
#pragma unroll
    for (int j = 0; j < 4; ++j) {
      const bf16* bp = vb0 + j * 16 * NPAD;
      bc[j] = ldpair(bp, bp + 8);
    }
#pragma unroll
    for (int kk = 0; kk < 7; ++kk) {
      v16bf an; v16bf bn[4];
      if (kk < 6) {
        an = ldpair(prow + (kk+1)*32, prow + (kk+1)*32 + 8);
#pragma unroll
        for (int j = 0; j < 4; ++j) {
          const bf16* bp = vb0 + j * 16 * NPAD + (kk+1)*32;
          bn[j] = ldpair(bp, bp + 8);
        }
      }
#pragma unroll
      for (int j = 0; j < 4; ++j) Oa[j] = wmma_bf16(ac, bc[j], Oa[j]);
      ac = an;
#pragma unroll
      for (int j = 0; j < 4; ++j) bc[j] = bn[j];
    }
#pragma unroll
    for (int j = 0; j < 4; ++j) {
      int d = j*16 + m;
#pragma unroll
      for (int r = 0; r < 8; ++r) {
        int rt = rbase + r + ksel*8;
        if (rt < NTOK)
          O[((size_t)win*NTOK + rt)*CD + head*HD + d] = (bf16)(Oa[j][r]);
      }
    }
  }
}

// ---------- kernel 5: proj GEMM + gamma1 residual (window->linear scatter) ----------
__global__ __launch_bounds__(256) void k_gemm_proj(const bf16* __restrict__ O,
                                                   const bf16* __restrict__ WT,
                                                   const float* __restrict__ bias,
                                                   const float* __restrict__ g1,
                                                   const float* __restrict__ xs,
                                                   float* __restrict__ xs2) {
  int wave = (blockIdx.x * blockDim.x + threadIdx.x) >> 5;
  const int cbs = CD / 64;
  if (wave >= (MTOK/32) * cbs) return;
  int rt = wave / cbs, cb = wave % cbs;
  size_t row0 = (size_t)rt * 32; int col0 = cb * 64;
  int lane = threadIdx.x & 31, m = lane & 15, ksel = lane >> 4;
  v8f accL[4] = {}, accH[4] = {};
  gemm_strip32(O, WT, CD, CD, row0, col0, CD, accL, accH);
  size_t tlin[16];
#pragma unroll
  for (int g = 0; g < 2; ++g)
#pragma unroll
  for (int r = 0; r < 8; ++r) {
    size_t twin = row0 + g*16 + r + ksel*8;
    int w = (int)(twin / NTOK), nl = (int)(twin % NTOK);
    int bi = w / 16, wq = w % 16, hb = wq / 4, wb = wq % 4;
    int i1 = nl / WSZ, j1 = nl % WSZ;
    tlin[g*8 + r] = (size_t)bi*3136 + (hb*WSZ + i1)*56 + wb*WSZ + j1;
  }
#pragma unroll
  for (int n = 0; n < 4; ++n) {
    int col = col0 + n*16 + m;
    float bj = bias[col], gj = g1[col];
#pragma unroll
    for (int r = 0; r < 8; ++r) {
      float vL = accL[n][r] + bj;
      xs2[tlin[r]*CD + col] = xs[tlin[r]*CD + col] + gj * vL;
      float vH = accH[n][r] + bj;
      xs2[tlin[8+r]*CD + col] = xs[tlin[8+r]*CD + col] + gj * vH;
    }
  }
}

// ---------- kernel 6: LN2 (linear order, contiguous) ----------
__global__ __launch_bounds__(256) void k_ln2(const float* __restrict__ xs2,
                                             const float* __restrict__ w,
                                             const float* __restrict__ b,
                                             bf16* __restrict__ h2) {
  int wave = (blockIdx.x * blockDim.x + threadIdx.x) >> 5;
  int lane = threadIdx.x & 31;
  if (wave >= MTOK) return;
  const float* p = xs2 + (size_t)wave * CD;
  float v[24], s = 0.f, s2 = 0.f;
#pragma unroll
  for (int i = 0; i < 24; ++i) { float f = p[lane + 32*i]; v[i] = f; s += f; s2 += f*f; }
#pragma unroll
  for (int mk = 16; mk >= 1; mk >>= 1) { s += __shfl_xor(s, mk); s2 += __shfl_xor(s2, mk); }
  float mu = s * (1.f/768.f);
  float rstd = rsqrtf(s2 * (1.f/768.f) - mu*mu + 1e-5f);
  bf16* ph = h2 + (size_t)wave * CD;
#pragma unroll
  for (int i = 0; i < 24; ++i) {
    int c = lane + 32*i;
    ph[c] = (bf16)((v[i]-mu)*rstd * w[c] + b[c]);
  }
}

// ---------- kernel 7: FC1 GEMM + exact GELU ----------
__global__ __launch_bounds__(256) void k_gemm_fc1(const bf16* __restrict__ h2,
                                                  const bf16* __restrict__ WT,
                                                  const float* __restrict__ bias,
                                                  bf16* __restrict__ a1) {
  int wave = (blockIdx.x * blockDim.x + threadIdx.x) >> 5;
  const int cbs = DFF / 64;
  if (wave >= (MTOK/32) * cbs) return;
  int rt = wave / cbs, cb = wave % cbs;
  size_t row0 = (size_t)rt * 32; int col0 = cb * 64;
  int lane = threadIdx.x & 31, m = lane & 15, ksel = lane >> 4;
  v8f accL[4] = {}, accH[4] = {};
  gemm_strip32(h2, WT, CD, CD, row0, col0, CD, accL, accH);
#pragma unroll
  for (int n = 0; n < 4; ++n) {
    int col = col0 + n*16 + m;
    float bj = bias[col];
#pragma unroll
    for (int r = 0; r < 8; ++r) {
      float vL = accL[n][r] + bj;
      float gL = 0.5f * vL * (1.0f + erff(vL * 0.70710678118654752f));
      a1[(row0 + r + ksel*8)*DFF + col] = (bf16)gL;
      float vH = accH[n][r] + bj;
      float gH = 0.5f * vH * (1.0f + erff(vH * 0.70710678118654752f));
      a1[(row0 + 16 + r + ksel*8)*DFF + col] = (bf16)gH;
    }
  }
}

// ---------- kernel 8: FC2 GEMM + gamma2 residual + BCHW scatter ----------
__global__ __launch_bounds__(256) void k_gemm_fc2(const bf16* __restrict__ a1,
                                                  const bf16* __restrict__ WT,
                                                  const float* __restrict__ bias,
                                                  const float* __restrict__ g2,
                                                  const float* __restrict__ xs2,
                                                  float* __restrict__ out) {
  int wave = (blockIdx.x * blockDim.x + threadIdx.x) >> 5;
  const int cbs = CD / 64;
  if (wave >= (MTOK/32) * cbs) return;
  int rt = wave / cbs, cb = wave % cbs;
  size_t row0 = (size_t)rt * 32; int col0 = cb * 64;
  int lane = threadIdx.x & 31, m = lane & 15, ksel = lane >> 4;
  v8f accL[4] = {}, accH[4] = {};
  gemm_strip32(a1, WT, DFF, DFF, row0, col0, DFF, accL, accH);
  int bi[16], y[16], xc[16];
#pragma unroll
  for (int g = 0; g < 2; ++g)
#pragma unroll
  for (int r = 0; r < 8; ++r) {
    int t = (int)(row0 + g*16 + r + ksel*8);
    bi[g*8+r] = t / 3136; int rem = t % 3136;
    y[g*8+r] = rem / 56; xc[g*8+r] = rem % 56;
  }
#pragma unroll
  for (int n = 0; n < 4; ++n) {
    int col = col0 + n*16 + m;
    float bj = bias[col], gj = g2[col];
#pragma unroll
    for (int r = 0; r < 8; ++r) {
      float vL = accL[n][r] + bj;
      float resL = xs2[(row0 + r + ksel*8)*CD + col] + gj * vL;
      out[(((size_t)bi[r]*CD + col)*56 + y[r])*56 + xc[r]] = resL;
      float vH = accH[n][r] + bj;
      float resH = xs2[(row0 + 16 + r + ksel*8)*CD + col] + gj * vH;
      out[(((size_t)bi[8+r]*CD + col)*56 + y[8+r])*56 + xc[8+r]] = resH;
    }
  }
}

// ---------- launcher ----------
extern "C" void kernel_launch(void* const* d_in, const int* in_sizes, int n_in,
                              void* d_out, int out_size, void* d_ws, size_t ws_size,
                              hipStream_t stream) {
  const float* x      = (const float*)d_in[0];
  const float* n1w    = (const float*)d_in[1];
  const float* n1b    = (const float*)d_in[2];
  const float* qkv_w  = (const float*)d_in[3];
  const float* qkv_b  = (const float*)d_in[4];
  const float* relt   = (const float*)d_in[5];
  const float* proj_w = (const float*)d_in[6];
  const float* proj_b = (const float*)d_in[7];
  const float* n2w    = (const float*)d_in[8];
  const float* n2b    = (const float*)d_in[9];
  const float* fc1_w  = (const float*)d_in[10];
  const float* fc1_b  = (const float*)d_in[11];
  const float* fc2_w  = (const float*)d_in[12];
  const float* fc2_b  = (const float*)d_in[13];
  const float* g1     = (const float*)d_in[14];
  const float* g2     = (const float*)d_in[15];
  float* out = (float*)d_out;

  char* ws = (char*)d_ws;
  size_t off = 0;
  auto carve = [&](size_t bytes) -> char* {
    char* p = ws + off; off += (bytes + 255) & ~(size_t)255; return p;
  };
  bf16* WT1 = (bf16*)carve((size_t)QKV_LD * CD * 2);
  bf16* WT2 = (bf16*)carve((size_t)CD * CD * 2);
  bf16* WT3 = (bf16*)carve((size_t)DFF * CD * 2);
  bf16* WT4 = (bf16*)carve((size_t)CD * DFF * 2);
  float* xs  = (float*)carve((size_t)MTOK * CD * 4);
  float* xs2 = (float*)carve((size_t)MTOK * CD * 4);
  bf16* h2   = (bf16*)carve((size_t)MTOK * CD * 2);
  // lifetime-overlapped region: [h | qkv], O aliases h, a1 aliases whole region
  char* region = carve((size_t)MTOK * CD * 2 + (size_t)MPAD * QKV_LD * 2);
  bf16* h   = (bf16*)region;
  bf16* Ob  = (bf16*)region;                       // written after h is dead
  bf16* qkv = (bf16*)(region + (size_t)MTOK * CD * 2);
  bf16* a1  = (bf16*)region;                       // written after qkv/O are dead

  // 1. weights -> bf16 transposed
  k_wT<<<(CD*QKV_LD + 255)/256, 256, 0, stream>>>(qkv_w,  WT1, CD,  QKV_LD);
  k_wT<<<(CD*CD     + 255)/256, 256, 0, stream>>>(proj_w, WT2, CD,  CD);
  k_wT<<<(CD*DFF    + 255)/256, 256, 0, stream>>>(fc1_w,  WT3, CD,  DFF);
  k_wT<<<(DFF*CD    + 255)/256, 256, 0, stream>>>(fc2_w,  WT4, DFF, CD);

  // 2. LN1 (one wave per token)
  k_ln1<<<(MTOK*32)/256, 256, 0, stream>>>(x, n1w, n1b, xs, h);

  // 3. QKV GEMM
  {
    int waves = (MTOK/32) * (QKV_LD/64);
    k_gemm_qkv<<<(waves + 7)/8, 256, 0, stream>>>(h, WT1, qkv_b, qkv);
  }

  // 4. attention: grid (head, window)
  k_attn<<<dim3(NHD, MTOK/NTOK), 128, 0, stream>>>(qkv, relt, Ob);

  // 5. proj + gamma1 residual
  {
    int waves = (MTOK/32) * (CD/64);
    k_gemm_proj<<<(waves + 7)/8, 256, 0, stream>>>(Ob, WT2, proj_b, g1, xs, xs2);
  }

  // 6. LN2
  k_ln2<<<(MTOK*32)/256, 256, 0, stream>>>(xs2, n2w, n2b, h2);

  // 7. FC1 + GELU
  {
    int waves = (MTOK/32) * (DFF/64);
    k_gemm_fc1<<<(waves + 7)/8, 256, 0, stream>>>(h2, WT3, fc1_b, a1);
  }

  // 8. FC2 + gamma2 residual + BCHW scatter
  {
    int waves = (MTOK/32) * (CD/64);
    k_gemm_fc2<<<(waves + 7)/8, 256, 0, stream>>>(a1, WT4, fc2_b, g2, xs2, out);
  }
}
